// PointNet2Encoder_59940563583542
// MI455X (gfx1250) — matrix operand order repaired and checked by reading, not compiled, not run on hardware
//
#include <hip/hip_runtime.h>
#include <cstdint>
#include <cstddef>

// ---------------- types ----------------
typedef _Float16 half_t;
typedef __attribute__((ext_vector_type(16))) _Float16 v16h;
typedef __attribute__((ext_vector_type(8)))  float    v8f;

// 128-bit chunk type for async global->LDS copies (matches builtin signature)
typedef __attribute__((__vector_size__(16))) int b128v;
typedef b128v __attribute__((address_space(1)))* g128p;
typedef b128v __attribute__((address_space(3)))* l128p;

// problem constants (from setup_inputs: B=8, N=2048)
constexpr int BN  = 2048;   // points per batch
constexpr int BM1 = 1024;   // fps stage 1 (ratio 0.5)
constexpr int BK1 = 32;     // max neighbors stage 1
constexpr int BM2 = 256;    // fps stage 2 (ratio 0.25 of M1)
constexpr int BK2 = 64;     // max neighbors stage 2

// ---------------- WMMA helpers ----------------
__device__ __forceinline__ v8f wmma_f16(v16h a, v16h b, v8f c) {
  // D = A(16x32 f16) * B(32x16 f16) + C(16x16 f32)
  return __builtin_amdgcn_wmma_f32_16x16x32_f16(
      false, a, false, b, (short)0, c, false, false);
}

// Slot (lane*16+e) inside a 16x32 A-fragment tile for element (row rr, k kk).
// Per ISA 7.12.2 16-bit A layout: lanes 0-15 hold K 0-7 / 16-23, lanes 16-31 hold K 8-15 / 24-31.
__device__ __forceinline__ int a_slot(int rr, int kk) {
  const int lane = rr + ((kk & 8) ? 16 : 0);
  const int e    = (kk & 7) + ((kk & 16) ? 8 : 0);
  return lane * 16 + e;
}

// B fragment from pre-swizzled weights: one contiguous 32B load per lane.
__device__ __forceinline__ v16h load_b_frag_swz(const half_t* __restrict__ wswz, int frag, int lane) {
  return *(const v16h*)(wswz + (((size_t)frag * 32 + lane) << 4));
}

__device__ __forceinline__ void atomicMaxFloat(float* addr, float v) {
  if (v >= 0.0f) atomicMax((int*)addr, __float_as_int(v));
  else           atomicMin((unsigned int*)addr, (unsigned int)__float_as_int(v));
}

// ---------------- kernel 0: init output to -inf ----------------
__global__ void init_out_kernel(float* out, int n) {
  int i = blockIdx.x * blockDim.x + threadIdx.x;
  if (i < n) out[i] = -INFINITY;
}

// ---------------- weight convert f32 -> f16, K-padded, B-fragment-swizzled ----------------
// dst layout: frag = kc*(N/16)+nt; element i = ((frag*32 + lane)*16 + e);
// value = W[kc*32 + (lane<16?0:8) + (e&7) + (e<8?0:16)][nt*16 + (lane&15)] (0 beyond Ksrc rows).
__global__ void convert_swz_kernel(const float* __restrict__ src, half_t* __restrict__ dst,
                                   int Ksrc, int Kdst, int N) {
  int i = blockIdx.x * blockDim.x + threadIdx.x;
  int total = Kdst * N;
  if (i >= total) return;
  int e      = i & 15;
  int lane   = (i >> 4) & 31;
  int frag   = i >> 9;
  int ntiles = N >> 4;
  int nt = frag % ntiles;
  int kc = frag / ntiles;
  int k  = kc * 32 + ((lane & 16) ? 8 : 0) + (e & 7) + ((e & 8) ? 16 : 0);
  int n  = nt * 16 + (lane & 15);
  float v = (k < Ksrc) ? src[k * N + n] : 0.0f;
  dst[i] = (half_t)v;
}

// ---------------- farthest point sampling ----------------
// One workgroup per batch; AoS points + running min-distance in LDS.
// Bulk global->LDS staging uses CDNA5 async-to-LDS when available (ASYNCcnt path).
template <int N, int M>
__global__ void fps_kernel(const float* __restrict__ pos,   // [B][N][3]
                           int* __restrict__ idx_out,       // [B][M]
                           float* __restrict__ pos_out) {   // [B][M][3]
  __shared__ __align__(16) float pts[N * 3];
  __shared__ float dist[N];
  __shared__ float redv[8];
  __shared__ int   redi[8];
  __shared__ int   curLds;
  const int b   = blockIdx.x;
  const int tid = threadIdx.x;
  const float* p = pos + (size_t)b * N * 3;

#if defined(__gfx1250__) && __has_builtin(__builtin_amdgcn_global_load_async_to_lds_b128) && __has_builtin(__builtin_amdgcn_s_wait_asynccnt)
  {
    constexpr int CHUNKS = (N * 3 * 4) / (256 * 16);
#pragma unroll
    for (int c = 0; c < CHUNKS; ++c) {
      const int off = (c * 256 + tid) * 16;
      __builtin_amdgcn_global_load_async_to_lds_b128(
          (g128p)((const char*)p + off),
          (l128p)((char*)pts + off),
          0, 0);
    }
    __builtin_amdgcn_s_wait_asynccnt(0);
  }
#else
  for (int i = tid; i < N * 3; i += 256) pts[i] = p[i];
#endif
  __syncthreads();

  const float p0x = pts[0], p0y = pts[1], p0z = pts[2];
  for (int i = tid; i < N; i += 256) {
    float dx = pts[i * 3 + 0] - p0x, dy = pts[i * 3 + 1] - p0y, dz = pts[i * 3 + 2] - p0z;
    dist[i] = dx * dx + dy * dy + dz * dz;
  }
  if (tid == 0) {
    idx_out[(size_t)b * M] = 0;
    pos_out[((size_t)b * M) * 3 + 0] = p0x;
    pos_out[((size_t)b * M) * 3 + 1] = p0y;
    pos_out[((size_t)b * M) * 3 + 2] = p0z;
    curLds = 0;
  }
  __syncthreads();
  int cur = 0;
  for (int j = 1; j < M; ++j) {
    const float cx = pts[cur * 3 + 0], cy = pts[cur * 3 + 1], cz = pts[cur * 3 + 2];
    float best = -1.0f;
    int   bi   = 0;
    for (int i = tid; i < N; i += 256) {
      float dx = pts[i * 3 + 0] - cx, dy = pts[i * 3 + 1] - cy, dz = pts[i * 3 + 2] - cz;
      float d  = dx * dx + dy * dy + dz * dz;
      float nd = fminf(dist[i], d);
      dist[i]  = nd;
      if (nd > best) { best = nd; bi = i; }
    }
#pragma unroll
    for (int off = 16; off > 0; off >>= 1) {
      float ov = __shfl_xor(best, off, 32);
      int   oi = __shfl_xor(bi,   off, 32);
      if (ov > best) { best = ov; bi = oi; }
    }
    if ((tid & 31) == 0) { redv[tid >> 5] = best; redi[tid >> 5] = bi; }
    __syncthreads();
    if (tid == 0) {
      float bv = redv[0]; int bx = redi[0];
#pragma unroll
      for (int w = 1; w < 8; ++w)
        if (redv[w] > bv) { bv = redv[w]; bx = redi[w]; }
      curLds = bx;
      idx_out[(size_t)b * M + j] = bx;
      pos_out[((size_t)b * M + j) * 3 + 0] = pts[bx * 3 + 0];
      pos_out[((size_t)b * M + j) * 3 + 1] = pts[bx * 3 + 1];
      pos_out[((size_t)b * M + j) * 3 + 2] = pts[bx * 3 + 2];
    }
    __syncthreads();
    cur = curLds;
  }
}

// ---------------- ball query: wave per centroid, ballot stream-compaction ----------------
template <int Npts, int Mc, int K>
__global__ void ballquery_kernel(const float* __restrict__ pts,
                                 const float* __restrict__ cent,
                                 int* __restrict__ nb,
                                 int* __restrict__ cnt_out,
                                 float r2) {
  const int wid  = (blockIdx.x * blockDim.x + threadIdx.x) >> 5;
  const int lane = threadIdx.x & 31;
  const int b = wid / Mc, m = wid % Mc;
  const float cx = cent[((size_t)b * Mc + m) * 3 + 0];
  const float cy = cent[((size_t)b * Mc + m) * 3 + 1];
  const float cz = cent[((size_t)b * Mc + m) * 3 + 2];
  const float* p = pts + (size_t)b * Npts * 3;
  int* nbp = nb + ((size_t)b * Mc + m) * K;
  int cnt = 0;
  for (int base = 0; base < Npts; base += 32) {
    if (cnt >= K) break;
    const int i = base + lane;
    float dx = p[i * 3 + 0] - cx, dy = p[i * 3 + 1] - cy, dz = p[i * 3 + 2] - cz;
    float d2 = dx * dx + dy * dy + dz * dz;
    bool in = d2 <= r2;
    unsigned mask = (unsigned)__ballot(in);
    int rank = __popc(mask & ((1u << lane) - 1u));
    int slot = cnt + rank;
    if (in && slot < K) nbp[slot] = i;
    cnt += __popc(mask);
  }
  if (lane == 0) cnt_out[(size_t)b * Mc + m] = (cnt < K) ? cnt : K;
}

// ---------------- SA1: wave per centroid: (32x[3->32]) -> 64 relu -> 128 -> masked max ----------------
// All LDS is wave-private; DS ops within a wave are in-order -> no block barriers needed.
__global__ void sa1_kernel(const float* __restrict__ pos,      // [B][N][3]
                           const float* __restrict__ cent,     // [B][M1][3]
                           const int* __restrict__ nb,         // [B][M1][K1]
                           const int* __restrict__ cnt_in,     // [B][M1]
                           const half_t* __restrict__ W1a_s,   // swizzled [1 kc][4 nt]
                           const float* __restrict__ b1a,      // [64]
                           const half_t* __restrict__ W1b_s,   // swizzled [2 kc][8 nt]
                           const float* __restrict__ b1b,      // [128]
                           half_t* __restrict__ x1_h) {        // [B][M1][128]
  __shared__ __align__(32) half_t Afrag[8][2][512];   // [wave][mt][lane*16+e]  16 KB
  __shared__ __align__(32) half_t Hfrag[8][4][512];   // [wave][mt*2+kc]        32 KB
  const int wave = threadIdx.x >> 5;
  const int lane = threadIdx.x & 31;
  const int c = blockIdx.x * 8 + wave;          // flat centroid id over B*M1
  const int b = c / BM1;
  const int cnt = cnt_in[c];
  const float cx = cent[(size_t)c * 3 + 0];
  const float cy = cent[(size_t)c * 3 + 1];
  const float cz = cent[(size_t)c * 3 + 2];
  const v8f  vzero{};
  const v16h hzero{};

  __builtin_prefetch(W1a_s, 0, 3);
  __builtin_prefetch(W1b_s, 0, 3);

  // zero A tiles, then scatter rel coords in A-fragment layout (lane == neighbor row)
  *(v16h*)&Afrag[wave][0][lane * 16] = hzero;
  *(v16h*)&Afrag[wave][1][lane * 16] = hzero;
  {
    float rx = 0.f, ry = 0.f, rz = 0.f;
    if (lane < cnt) {
      int i = nb[(size_t)c * BK1 + lane];
      const float* pp = pos + ((size_t)b * BN + i) * 3;
      rx = pp[0] - cx; ry = pp[1] - cy; rz = pp[2] - cz;
    }
    half_t* tile = &Afrag[wave][lane >> 4][0];
    const int rr = lane & 15;
    tile[a_slot(rr, 0)] = (half_t)rx;
    tile[a_slot(rr, 1)] = (half_t)ry;
    tile[a_slot(rr, 2)] = (half_t)rz;
  }

  const v16h a0 = *(const v16h*)&Afrag[wave][0][lane * 16];
  const v16h a1 = *(const v16h*)&Afrag[wave][1][lane * 16];

  // GEMM1: hidden[32][64] = relu(A @ W1a + b1a) -> Hfrag (A-fragment layout for GEMM2)
#pragma unroll
  for (int nt = 0; nt < 4; ++nt) {
    v16h bf = load_b_frag_swz(W1a_s, nt, lane);
    v8f h0 = wmma_f16(a0, bf, vzero);
    v8f h1 = wmma_f16(a1, bf, vzero);
    const float bias = b1a[nt * 16 + (lane & 15)];
    const int kk  = (nt & 1) * 16 + (lane & 15);
    const int kc2 = nt >> 1;
#pragma unroll
    for (int r = 0; r < 8; ++r) {
      const int rr = (lane < 16) ? r : (r + 8);
      Hfrag[wave][0 * 2 + kc2][a_slot(rr, kk)] = (half_t)fmaxf(h0[r] + bias, 0.f);
      Hfrag[wave][1 * 2 + kc2][a_slot(rr, kk)] = (half_t)fmaxf(h1[r] + bias, 0.f);
    }
  }

  // GEMM2: out[32][128] = hidden @ W1b + b1b, masked max over neighbor rows -> x1
#pragma unroll
  for (int nt = 0; nt < 8; ++nt) {
    v8f acc0 = vzero, acc1 = vzero;
#pragma unroll
    for (int kc = 0; kc < 2; ++kc) {
      v16h bf = load_b_frag_swz(W1b_s, kc * 8 + nt, lane);
      v16h ha0 = *(const v16h*)&Hfrag[wave][0 * 2 + kc][lane * 16];
      v16h ha1 = *(const v16h*)&Hfrag[wave][1 * 2 + kc][lane * 16];
      acc0 = wmma_f16(ha0, bf, acc0);
      acc1 = wmma_f16(ha1, bf, acc1);
    }
    const float bias = b1b[nt * 16 + (lane & 15)];
    float best = -INFINITY;
#pragma unroll
    for (int r = 0; r < 8; ++r) {
      const int row0 = (lane < 16) ? r : (r + 8);
      if (row0 < cnt)      best = fmaxf(best, acc0[r] + bias);
      if (row0 + 16 < cnt) best = fmaxf(best, acc1[r] + bias);
    }
    best = fmaxf(best, __shfl_xor(best, 16, 32));
    if (lane < 16)
      x1_h[(size_t)c * 128 + nt * 16 + lane] = (half_t)best;
  }
}

// ---------------- SA2: block per centroid: (64x[131->160]) -> 256 relu -> 512 -> max -> atomic ----------------
__global__ void sa2_kernel(const float* __restrict__ pos1,     // [B][M1][3]
                           const float* __restrict__ cent,     // [B][M2][3]
                           const int* __restrict__ nb,         // [B][M2][K2]
                           const int* __restrict__ cnt_in,     // [B][M2]
                           const half_t* __restrict__ x1_h,    // [B][M1][128]
                           const half_t* __restrict__ W2a_s,   // swizzled [5 kc][16 nt]
                           const float* __restrict__ b2a,      // [256]
                           const half_t* __restrict__ W2b_s,   // swizzled [8 kc][32 nt]
                           const float* __restrict__ b2b,      // [512]
                           float* __restrict__ out) {          // [B][512]
  __shared__ __align__(32) half_t Afrag[20][512];   // tile = mt*5+kc, 20 KB
  __shared__ __align__(32) half_t Hfrag[32][512];   // tile = mt*8+kc, 32 KB
  const int c = blockIdx.x;        // flat over B*M2
  const int b = c / BM2;
  const int cnt = cnt_in[c];
  const int tid  = threadIdx.x;
  const int wave = tid >> 5;
  const int lane = tid & 31;
  const float cx = cent[(size_t)c * 3 + 0];
  const float cy = cent[(size_t)c * 3 + 1];
  const float cz = cent[(size_t)c * 3 + 2];
  const v8f  vzero{};
  const v16h hzero{};

  __builtin_prefetch(W2a_s, 0, 3);
  __builtin_prefetch(W2b_s, 0, 3);

  // zero all A tiles
  for (int t = wave; t < 20; t += 8)
    *(v16h*)&Afrag[t][lane * 16] = hzero;
  __syncthreads();

  // scatter features into A-fragment layout: feat = [x1(128) | rel(3)] (rest stays 0)
  {
    const int row  = tid >> 2;     // 4 threads per neighbor row
    const int part = tid & 3;      // 33 cols each (132 >= 131)
    int nbr = -1;
    if (row < cnt) nbr = nb[(size_t)c * BK2 + row];
    if (nbr >= 0) {
      const float* pp = pos1 + ((size_t)b * BM1 + nbr) * 3;
      const float rel[3] = { pp[0] - cx, pp[1] - cy, pp[2] - cz };
      const half_t* xr = x1_h + ((size_t)b * BM1 + nbr) * 128;
      const int mt = row >> 4, rr = row & 15;
      for (int j = 0; j < 33; ++j) {
        const int col = part * 33 + j;
        if (col >= 131) break;
        half_t v = (col < 128) ? xr[col] : (half_t)rel[col - 128];
        Afrag[mt * 5 + (col >> 5)][a_slot(rr, col & 31)] = v;
      }
    }
  }
  __syncthreads();

  // GEMM1: hidden[64][256] = relu(A @ W2a + b2a); wave owns col-tiles {w, w+8}, all 4 row-tiles
#pragma unroll
  for (int t = 0; t < 2; ++t) {
    const int nt = wave + 8 * t;
    v8f acc[4];
#pragma unroll
    for (int mt = 0; mt < 4; ++mt) acc[mt] = vzero;
#pragma unroll
    for (int kc = 0; kc < 5; ++kc) {
      v16h bf = load_b_frag_swz(W2a_s, kc * 16 + nt, lane);
#pragma unroll
      for (int mt = 0; mt < 4; ++mt) {
        v16h a = *(const v16h*)&Afrag[mt * 5 + kc][lane * 16];
        acc[mt] = wmma_f16(a, bf, acc[mt]);
      }
    }
    const float bias = b2a[nt * 16 + (lane & 15)];
    const int kk  = (nt & 1) * 16 + (lane & 15);
    const int kc2 = nt >> 1;
#pragma unroll
    for (int mt = 0; mt < 4; ++mt) {
#pragma unroll
      for (int r = 0; r < 8; ++r) {
        const int rr = (lane < 16) ? r : (r + 8);
        Hfrag[mt * 8 + kc2][a_slot(rr, kk)] = (half_t)fmaxf(acc[mt][r] + bias, 0.f);
      }
    }
  }
  __syncthreads();

  // GEMM2 + masked max over 64 rows; wave owns col-tiles {w, w+8, w+16, w+24}
#pragma unroll
  for (int t = 0; t < 4; ++t) {
    const int nt = wave + 8 * t;
    v8f acc[4];
#pragma unroll
    for (int mt = 0; mt < 4; ++mt) acc[mt] = vzero;
#pragma unroll
    for (int kc = 0; kc < 8; ++kc) {
      v16h bf = load_b_frag_swz(W2b_s, kc * 32 + nt, lane);
#pragma unroll
      for (int mt = 0; mt < 4; ++mt) {
        v16h a = *(const v16h*)&Hfrag[mt * 8 + kc][lane * 16];
        acc[mt] = wmma_f16(a, bf, acc[mt]);
      }
    }
    const float bias = b2b[nt * 16 + (lane & 15)];
    float best = -INFINITY;
#pragma unroll
    for (int mt = 0; mt < 4; ++mt) {
#pragma unroll
      for (int r = 0; r < 8; ++r) {
        const int row = mt * 16 + ((lane < 16) ? r : (r + 8));
        if (row < cnt) best = fmaxf(best, acc[mt][r] + bias);
      }
    }
    best = fmaxf(best, __shfl_xor(best, 16, 32));
    if (lane < 16)
      atomicMaxFloat(&out[(size_t)b * 512 + nt * 16 + lane], best);
  }
}

// ---------------- launcher ----------------
extern "C" void kernel_launch(void* const* d_in, const int* in_sizes, int n_in,
                              void* d_out, int out_size, void* d_ws, size_t ws_size,
                              hipStream_t stream) {
  const float* pos = (const float*)d_in[0];
  const float* W1a = (const float*)d_in[1];
  const float* b1a = (const float*)d_in[2];
  const float* W1b = (const float*)d_in[3];
  const float* b1b = (const float*)d_in[4];
  const float* W2a = (const float*)d_in[5];
  const float* b2a = (const float*)d_in[6];
  const float* W2b = (const float*)d_in[7];
  const float* b2b = (const float*)d_in[8];
  float* out = (float*)d_out;

  const int B = in_sizes[0] / (BN * 3);   // 8

  char* ws = (char*)d_ws;
  int*    idx1 = (int*)   (ws + 0);         // B*M1 ints
  float*  pos1 = (float*) (ws + 32768);     // B*M1*3 f32
  int*    nb1  = (int*)   (ws + 131072);    // B*M1*K1 ints
  int*    cnt1 = (int*)   (ws + 1179648);   // B*M1 ints
  half_t* x1h  = (half_t*)(ws + 1212416);   // B*M1*128 f16
  int*    idx2 = (int*)   (ws + 3309568);   // B*M2 ints
  float*  pos2 = (float*) (ws + 3317760);   // B*M2*3 f32
  int*    nb2  = (int*)   (ws + 3342336);   // B*M2*K2 ints
  int*    cnt2 = (int*)   (ws + 3866624);   // B*M2 ints
  half_t* W1as = (half_t*)(ws + 3874816);   // [32][64]   swizzled
  half_t* W1bs = (half_t*)(ws + 3878912);   // [64][128]  swizzled
  half_t* W2as = (half_t*)(ws + 3895296);   // [160][256] swizzled
  half_t* W2bs = (half_t*)(ws + 3977216);   // [256][512] swizzled

  init_out_kernel<<<(B * 512 + 255) / 256, 256, 0, stream>>>(out, B * 512);
  convert_swz_kernel<<<(32 * 64   + 255) / 256, 256, 0, stream>>>(W1a, W1as,   3,  32,  64);
  convert_swz_kernel<<<(64 * 128  + 255) / 256, 256, 0, stream>>>(W1b, W1bs,  64,  64, 128);
  convert_swz_kernel<<<(160 * 256 + 255) / 256, 256, 0, stream>>>(W2a, W2as, 131, 160, 256);
  convert_swz_kernel<<<(256 * 512 + 255) / 256, 256, 0, stream>>>(W2b, W2bs, 256, 256, 512);

  fps_kernel<BN, BM1><<<B, 256, 0, stream>>>(pos, idx1, pos1);
  ballquery_kernel<BN, BM1, BK1><<<(B * BM1) / 8, 256, 0, stream>>>(pos, pos1, nb1, cnt1, 0.2f * 0.2f);
  sa1_kernel<<<(B * BM1) / 8, 256, 0, stream>>>(pos, pos1, nb1, cnt1, W1as, b1a, W1bs, b1b, x1h);

  fps_kernel<BM1, BM2><<<B, 256, 0, stream>>>(pos1, idx2, pos2);
  ballquery_kernel<BM1, BM2, BK2><<<(B * BM2) / 8, 256, 0, stream>>>(pos1, pos2, nb2, cnt2, 0.4f * 0.4f);
  sa2_kernel<<<B * BM2, 256, 0, stream>>>(pos1, pos2, nb2, cnt2, x1h, W2as, b2a, W2bs, b2b, out);
}
